// PCFG_62912680952456
// MI455X (gfx1250) — matrix operand order repaired and checked by reading, not compiled, not run on hardware
//
#include <hip/hip_runtime.h>
#include <hip/hip_bf16.h>

// PCFG inside algorithm for MI455X (gfx1250, wave32, WMMA).
// Dims fixed by the reference: B=64, n=50, nt=30, t=60, S=90 (padded to 96).
#define B_     64
#define N_     50
#define NT_    30
#define T_     60
#define S_     90
#define SP_    96      // S padded to multiple of 16/32
#define NP1_   51      // n+1
#define KPADMX 64      // max splits (49) padded to multiple of 32
#define NEGV   (-1000000000.0f)
#define NINF   (-3.0e38f)

typedef __attribute__((ext_vector_type(16))) __bf16          v16bf;
typedef __attribute__((ext_vector_type(8)))  __bf16          v8bf;
typedef __attribute__((ext_vector_type(8)))  float           v8f;
typedef __attribute__((ext_vector_type(4)))  unsigned int    u32x4;

__device__ __forceinline__ unsigned short f2bf(float f) {
  unsigned int u = __builtin_bit_cast(unsigned int, f);
  u += 0x7FFFu + ((u >> 16) & 1u);          // round-to-nearest-even
  return (unsigned short)(u >> 16);
}
// bf16 pair unpack from a packed u32 (no cvt instructions needed)
__device__ __forceinline__ float bflo(unsigned int u) {
  return __builtin_bit_cast(float, u << 16);
}
__device__ __forceinline__ float bfhi(unsigned int u) {
  return __builtin_bit_cast(float, u & 0xffff0000u);
}

// ---------------------------------------------------------------------------
// Kernel 0: beta = NEG everywhere; diagonal beta[b,i,i+1,nt:S] = unary[b,i,:]
// Padding states [90,96) stay NEG so exp() of them is exactly 0 downstream.
// ---------------------------------------------------------------------------
__global__ void pcfg_init_beta(float* __restrict__ beta,
                               const float* __restrict__ unary) {
  size_t idx = (size_t)blockIdx.x * blockDim.x + threadIdx.x;
  const size_t tot = (size_t)B_ * NP1_ * NP1_ * SP_;
  if (idx >= tot) return;
  int st = (int)(idx % SP_);
  size_t r = idx / SP_;
  int j = (int)(r % NP1_); r /= NP1_;
  int i = (int)(r % NP1_);
  int b = (int)(r / NP1_);
  float v = NEGV;
  if (j == i + 1 && i < N_ && st >= NT_ && st < S_)
    v = unary[((size_t)b * N_ + i) * T_ + (st - NT_)];
  beta[idx] = v;
}

// ---------------------------------------------------------------------------
// Kernel 1: per (b,p): m_rule = max over 90x90; erule = bf16(exp(rule - m)),
// zero-padded to 96x96 so padded rows/cols of `outer` contribute nothing.
// ---------------------------------------------------------------------------
__global__ __launch_bounds__(256)
void pcfg_prep_rules(const float* __restrict__ rules,
                     unsigned short* __restrict__ erule,
                     float* __restrict__ m_rule) {
  const int bp   = blockIdx.x;            // b*NT_ + p
  const int tid  = threadIdx.x;
  const int wv   = tid >> 5;
  const int lane = tid & 31;
  const float* src = rules + (size_t)bp * (S_ * S_);
  __shared__ float red[8];
  float m = NINF;
  for (int q = tid; q < S_ * S_; q += 256) m = fmaxf(m, src[q]);
  for (int o = 16; o; o >>= 1) m = fmaxf(m, __shfl_xor(m, o, 32));
  if (lane == 0) red[wv] = m;
  __syncthreads();
  if (tid == 0) {
    float mm = red[0];
    for (int i = 1; i < 8; ++i) mm = fmaxf(mm, red[i]);
    red[0] = mm;
    m_rule[bp] = mm;
  }
  __syncthreads();
  m = red[0];
  unsigned short* dst = erule + (size_t)bp * (SP_ * SP_);
  for (int q = tid; q < SP_ * SP_; q += 256) {
    int l = q / SP_, r = q % SP_;
    dst[q] = (l < S_ && r < S_) ? f2bf(__expf(src[l * S_ + r] - m))
                                : (unsigned short)0;
  }
}

// ---------------------------------------------------------------------------
// Kernel 2: one block per (b, span) for a given width w. 8 wave32s.
//  A: per-split maxes + scaled exponentials -> LDS (bf16). BOTH el and er are
//     stored K-major ([state][k]) so each WMMA fragment is a contiguous LDS
//     vector load (ds_load_b128), no per-element assembly.
//  B: outer[l,r] = sum_k el[k,l]*er[k,r] via v_wmma_f32_16x16x32_bf16,
//     36 16x16 tiles of the 96x96 result; K-chunks unrolled (Kp = 32 or 64).
//  C: ssum[p] = <erule[b,p,:,:], outer>, 128-bit loads; beta = log + M + m_rule.
// ---------------------------------------------------------------------------
__global__ __launch_bounds__(256)
void pcfg_span_kernel(float* __restrict__ beta,
                      const unsigned short* __restrict__ erule,
                      const float* __restrict__ m_rule,
                      int w) {
  const int s    = blockIdx.x;
  const int b    = blockIdx.y;
  const int K    = w - 1;
  const int Kp   = (K + 31) & ~31;         // 32 or 64
  const int tid  = threadIdx.x;
  const int wv   = tid >> 5;
  const int lane = tid & 31;

  __shared__ __align__(32) unsigned short sh_elt[SP_ * KPADMX]; // el^T: [l][k]
  __shared__ __align__(32) unsigned short sh_ert[SP_ * KPADMX]; // er^T: [r][k]
  __shared__ __align__(16) unsigned short sh_out[SP_ * SP_];    // outer [l][r]
  __shared__ float sh_ml[KPADMX], sh_mr[KPADMX];
  __shared__ float sh_M;

  // ---- Phase A1: per-split row maxes (one wave per split, strided)
  for (int k = wv; k < K; k += 8) {
    const int u = s + 1 + k;
    const float* lc = beta + ((size_t)((b * NP1_ + s) * NP1_ + u)) * SP_;
    const float* rc = beta + ((size_t)((b * NP1_ + u) * NP1_ + (s + w))) * SP_;
    float ml = NINF, mr = NINF;
    for (int j = lane; j < SP_; j += 32) {
      ml = fmaxf(ml, lc[j]);
      mr = fmaxf(mr, rc[j]);
    }
    for (int o = 16; o; o >>= 1) {
      ml = fmaxf(ml, __shfl_xor(ml, o, 32));
      mr = fmaxf(mr, __shfl_xor(mr, o, 32));
    }
    if (lane == 0) { sh_ml[k] = ml; sh_mr[k] = mr; }
  }
  __syncthreads();
  if (wv == 0) {                            // M = max_k (ml+mr)
    float v = NINF;
    for (int k = lane; k < K; k += 32) v = fmaxf(v, sh_ml[k] + sh_mr[k]);
    for (int o = 16; o; o >>= 1) v = fmaxf(v, __shfl_xor(v, o, 32));
    if (lane == 0) sh_M = v;
  }
  __syncthreads();
  const float M = sh_M;

  // ---- Phase A2: exponentials into LDS as bf16 (K-major); zero K-padding
  for (int k = wv; k < Kp; k += 8) {
    if (k < K) {
      const int u = s + 1 + k;
      const float* lc = beta + ((size_t)((b * NP1_ + s) * NP1_ + u)) * SP_;
      const float* rc = beta + ((size_t)((b * NP1_ + u) * NP1_ + (s + w))) * SP_;
      const float ml = sh_ml[k], mr = sh_mr[k];
      const float scale = __expf(ml + mr - M);
      for (int j = lane; j < SP_; j += 32) {
        sh_elt[j * KPADMX + k] = f2bf(__expf(lc[j] - ml) * scale);
        sh_ert[j * KPADMX + k] = f2bf(__expf(rc[j] - mr));
      }
    } else {
      for (int j = lane; j < SP_; j += 32) {
        sh_elt[j * KPADMX + k] = 0;
        sh_ert[j * KPADMX + k] = 0;
      }
    }
  }
  __syncthreads();

  // ---- Phase B: outer = el^T x er via WMMA bf16 (f32 accumulate)
  // Lane layouts per CDNA5 ISA 7.12.2 (16-bit operands, wave32):
  //  A 16x32: lane row M=lane&15; elems 0-7 -> K=h*8+0..7, 8-15 -> K=16+h*8+..
  //  B 32x16: lane col N=lane&15; elems 0-7 -> K=h*8+0..7, 8-15 -> K=16+h*8+..
  //  C/D f32: VGPR r -> row r + 8*h, col = lane&15
  const int half = lane >> 4;
  const int mn   = lane & 15;
  for (int t = wv; t < 36; t += 8) {
    const int tl = t / 6, tr = t % 6;
    const unsigned short* arow = &sh_elt[(tl * 16 + mn) * KPADMX];
    const unsigned short* brow = &sh_ert[(tr * 16 + mn) * KPADMX];
    v8f acc = {0.f, 0.f, 0.f, 0.f, 0.f, 0.f, 0.f, 0.f};
    {   // K chunk 0 (always present)
      v8bf a0 = *(const v8bf*)(arow + half * 8);
      v8bf a1 = *(const v8bf*)(arow + 16 + half * 8);
      v16bf af = __builtin_shufflevector(a0, a1,
                   0,1,2,3,4,5,6,7,8,9,10,11,12,13,14,15);
      v8bf b0 = *(const v8bf*)(brow + half * 8);
      v8bf b1 = *(const v8bf*)(brow + 16 + half * 8);
      v16bf bf = __builtin_shufflevector(b0, b1,
                   0,1,2,3,4,5,6,7,8,9,10,11,12,13,14,15);
      acc = __builtin_amdgcn_wmma_f32_16x16x32_bf16(
                false, af, false, bf, (short)0, acc, false, false);
    }
    if (Kp > 32) {  // K chunk 1 (widths >= 34)
      v8bf a0 = *(const v8bf*)(arow + 32 + half * 8);
      v8bf a1 = *(const v8bf*)(arow + 48 + half * 8);
      v16bf af = __builtin_shufflevector(a0, a1,
                   0,1,2,3,4,5,6,7,8,9,10,11,12,13,14,15);
      v8bf b0 = *(const v8bf*)(brow + 32 + half * 8);
      v8bf b1 = *(const v8bf*)(brow + 48 + half * 8);
      v16bf bf = __builtin_shufflevector(b0, b1,
                   0,1,2,3,4,5,6,7,8,9,10,11,12,13,14,15);
      acc = __builtin_amdgcn_wmma_f32_16x16x32_bf16(
                false, af, false, bf, (short)0, acc, false, false);
    }
#pragma unroll
    for (int r = 0; r < 8; ++r) {
      sh_out[(tl * 16 + r + half * 8) * SP_ + tr * 16 + mn] = f2bf(acc[r]);
    }
  }
  __syncthreads();

  // ---- Phase C: contract erule with outer; one wave per nonterminal p.
  // 128-bit global + LDS loads, 8 bf16 MACs per load pair per lane.
  for (int p = wv; p < NT_; p += 8) {
    const unsigned short* er_p = erule + ((size_t)(b * NT_ + p)) * (SP_ * SP_);
    __builtin_prefetch(er_p + lane * 64, 0, 1);   // global_prefetch_b8
    float acc = 0.f;
#pragma unroll 4
    for (int it = 0; it < (SP_ * SP_) / (32 * 8); ++it) {   // 36 iters
      const int base = (it * 32 + lane) * 8;
      u32x4 ea = *(const u32x4*)(er_p + base);
      u32x4 ob = *(const u32x4*)(sh_out + base);
#pragma unroll
      for (int c = 0; c < 4; ++c) {
        acc = fmaf(bflo(ea[c]), bflo(ob[c]), acc);
        acc = fmaf(bfhi(ea[c]), bfhi(ob[c]), acc);
      }
    }
    for (int o = 16; o; o >>= 1) acc += __shfl_xor(acc, o, 32);
    if (lane == 0) {
      beta[((size_t)((b * NP1_ + s) * NP1_ + (s + w))) * SP_ + p] =
          __logf(acc) + M + m_rule[b * NT_ + p];
    }
  }
}

// ---------------------------------------------------------------------------
// Kernel 3: log_Z[b] = logsumexp_p(beta[b,0,n,p] + root[b,p]); one wave per b
// ---------------------------------------------------------------------------
__global__ __launch_bounds__(256)
void pcfg_root(const float* __restrict__ beta,
               const float* __restrict__ root,
               float* __restrict__ out) {
  const int wv   = threadIdx.x >> 5;
  const int lane = threadIdx.x & 31;
  const int b    = blockIdx.x * 8 + wv;
  if (b >= B_) return;
  float v = NINF;
  if (lane < NT_)
    v = beta[((size_t)((b * NP1_ + 0) * NP1_ + N_)) * SP_ + lane] +
        root[b * NT_ + lane];
  float m = v;
  for (int o = 16; o; o >>= 1) m = fmaxf(m, __shfl_xor(m, o, 32));
  float e = (lane < NT_) ? __expf(v - m) : 0.f;
  for (int o = 16; o; o >>= 1) e += __shfl_xor(e, o, 32);
  if (lane == 0) out[b] = m + __logf(e);
}

// ---------------------------------------------------------------------------
extern "C" void kernel_launch(void* const* d_in, const int* in_sizes, int n_in,
                              void* d_out, int out_size, void* d_ws, size_t ws_size,
                              hipStream_t stream) {
  const float* unary = (const float*)d_in[0];   // (B, n, t)
  const float* rules = (const float*)d_in[1];   // (B, nt, S, S)
  const float* root  = (const float*)d_in[2];   // (B, nt)
  // d_in[3]=nt(30), d_in[4]=t(60): fixed by reference shapes, baked in.

  // Workspace partition (~100 MB; resident in MI455X's 192 MB L2):
  float* beta = (float*)d_ws;                                   // 64 MB f32
  const size_t betaN  = (size_t)B_ * NP1_ * NP1_ * SP_;
  unsigned short* erule = (unsigned short*)(beta + betaN);      // 35 MB bf16
  const size_t eruleN = (size_t)B_ * NT_ * SP_ * SP_;
  float* m_rule = (float*)(erule + eruleN);                     // 7.5 KB

  {
    const size_t tot = betaN;
    const int blocks = (int)((tot + 255) / 256);
    pcfg_init_beta<<<blocks, 256, 0, stream>>>(beta, unary);
  }
  pcfg_prep_rules<<<B_ * NT_, 256, 0, stream>>>(rules, erule, m_rule);

  for (int w = 2; w <= N_; ++w) {
    dim3 grid(N_ - w + 1, B_);
    pcfg_span_kernel<<<grid, 256, 0, stream>>>(beta, erule, m_rule, w);
  }

  pcfg_root<<<(B_ + 7) / 8, 256, 0, stream>>>(beta, root, (float*)d_out);
}
